// SelfAttention_68788196212933
// MI455X (gfx1250) — matrix operand order repaired and checked by reading, not compiled, not run on hardware
//
#include <hip/hip_runtime.h>
#include <math.h>

typedef __attribute__((ext_vector_type(16))) _Float16 v16h;
typedef __attribute__((ext_vector_type(8)))  _Float16 v8h;
typedef __attribute__((ext_vector_type(8)))  float    v8f;
typedef __attribute__((ext_vector_type(4)))  unsigned int v4u;
typedef __attribute__((ext_vector_type(8)))  int      v8i;
typedef __attribute__((ext_vector_type(4)))  int      v4i;
typedef _Float16 h16;

#define Bb 4
#define Ss 4096
#define Ee 1024
#define Dd 64
#define NWELEM (Dd * Ee)   // 65536 elements per weight matrix

#if __has_builtin(__builtin_amdgcn_tensor_load_to_lds)
#define USE_TDM 1
#else
#define USE_TDM 0
#endif

// ---------------------------------------------------------------------------
// Kernel 0: convert W_q|W_k|W_v (f32 [64,1024]) to f16 once (384 KB, L2-hot).
// ---------------------------------------------------------------------------
__global__ __launch_bounds__(256) void cvt_w_kernel(
    const float* __restrict__ wq, const float* __restrict__ wk,
    const float* __restrict__ wv, h16* __restrict__ wh)
{
  const int idx = (int)(blockIdx.x * 256 + threadIdx.x);   // 0 .. 3*65536-1
  const int mat = idx >> 16, off = idx & 0xFFFF;
  const float* src = (mat == 0) ? wq : (mat == 1) ? wk : wv;
  wh[idx] = (h16)src[off];
}

// ---------------------------------------------------------------------------
// Kernel 1: fused Q/K/V projection, one wave per 16x64 tile, f16 WMMA / f32
// accumulate. Q/K stored row-major f16; V stored transposed (Vt[b][d][s]).
// ---------------------------------------------------------------------------
__global__ __launch_bounds__(256) void proj_kernel(
    const float* __restrict__ Xq, const float* __restrict__ Xk, const float* __restrict__ Xv,
    const h16* __restrict__ Wh,
    h16* __restrict__ Qh, h16* __restrict__ Kh, h16* __restrict__ Vt)
{
  const int lane = (int)(threadIdx.x & 31);
  const int wave = (int)((blockIdx.x * blockDim.x + threadIdx.x) >> 5); // 0..3071
  const int mat  = wave >> 10;      // 0=Q 1=K 2=V
  const int tile = wave & 1023;     // 16-row tile over B*S = 16384 rows
  const float* X = (mat == 0) ? Xq : (mat == 1) ? Xk : Xv;
  const h16*   W = Wh + (size_t)mat * NWELEM;

  const int m    = lane & 15;
  const int kg   = lane >> 4;
  const int row0 = tile * 16;

  v8f acc[4] = {};
  const float* arowBase = X + (size_t)(row0 + m) * Ee + kg * 16;

  for (int e0 = 0; e0 < Ee; e0 += 32) {
    v16h a;
    const float* ar = arowBase + e0;
    #pragma unroll
    for (int j = 0; j < 8; ++j) { a[j] = (h16)ar[j]; a[8 + j] = (h16)ar[8 + j]; }
    #pragma unroll
    for (int n4 = 0; n4 < 4; ++n4) {
      const v16h bf = *(const v16h*)(W + (size_t)(n4 * 16 + m) * Ee + e0 + kg * 16);
      acc[n4] = __builtin_amdgcn_wmma_f32_16x16x32_f16(
          false, a, false, bf, (short)0, acc[n4], false, false);
    }
  }

  if (mat < 2) {
    h16* Y = (mat == 0) ? Qh : Kh;
    #pragma unroll
    for (int n4 = 0; n4 < 4; ++n4)
      #pragma unroll
      for (int r = 0; r < 8; ++r)
        Y[(size_t)(row0 + r + 8 * kg) * Dd + n4 * 16 + m] = (h16)acc[n4][r];
  } else {
    const int bidx = row0 >> 12;          // row0 / S
    const int s0   = (row0 & (Ss - 1)) + 8 * kg;
    #pragma unroll
    for (int n4 = 0; n4 < 4; ++n4) {
      v8h pk;
      #pragma unroll
      for (int r = 0; r < 8; ++r) pk[r] = (h16)acc[n4][r];
      *(v8h*)(Vt + ((size_t)bidx * Dd + n4 * 16 + m) * Ss + s0) = pk;
    }
  }
}

#if USE_TDM
// Issue a 2D TDM tile load (16-bit elements) per CDNA5 ISA D# groups 0/1
// (8.3/8.4): count=1, type=2 ("image"), data_size=1 (2 bytes), groups 2/3
// and the trailing group zero (2D tensor: tile_dim2/3/4 = 0 -> unused).
// All operands are uniform. clang-23 toolchain: 6-arg builtin
// (v4u g0, v8i g1, v4i, v4i, v8i, i32 cpol).
__device__ __forceinline__ void tdm_load_tile_b16(
    unsigned lds_byte_off, const h16* gptr,
    unsigned tile_d0, unsigned tile_d1,
    unsigned tensor_d0, unsigned tensor_d1, unsigned stride0_elems)
{
  unsigned long long ga = (unsigned long long)(uintptr_t)gptr;
  v4u g0;
  g0[0] = 1u;                                                // count=1, user mode
  g0[1] = lds_byte_off;                                      // lds_addr (bytes)
  g0[2] = (unsigned)ga;                                      // global_addr[31:0]
  g0[3] = (unsigned)((ga >> 32) & 0x01FFFFFFu) | (2u << 30); // addr[56:32] | type=2
  v8i g1;
  g1[0] = (int)(1u << 16);                                   // data_size = 2B
  g1[1] = (int)((tensor_d0 & 0xFFFFu) << 16);                // tensor_dim0 lo
  g1[2] = (int)((tensor_d0 >> 16) | ((tensor_d1 & 0xFFFFu) << 16));
  g1[3] = (int)((tensor_d1 >> 16) | (tile_d0 << 16));        // tile_dim0
  g1[4] = (int)(tile_d1 & 0xFFFFu);                          // tile_dim1 (dim2=0)
  g1[5] = (int)stride0_elems;                                // tensor_dim0_stride
  g1[6] = 0;
  g1[7] = 0;
  const v4i gz4 = {0, 0, 0, 0};
  const v8i gz8 = {0, 0, 0, 0, 0, 0, 0, 0};
  __builtin_amdgcn_tensor_load_to_lds(g0, g1, gz4, gz4, gz8, 0);
}
#endif

// ---------------------------------------------------------------------------
// Kernel 2: flash attention. Block = 4 waves = 64 query rows (256 blocks to
// fill the machine); 32-key K/V tiles staged in double-buffered LDS, filled by
// the Tensor Data Mover (lead wave issues D#, waits TENSORcnt, one barrier per
// iteration). Fallback: manual cooperative staging. Online softmax in log2
// domain; P reshaped C-layout -> A-fragment via per-wave LDS (in-order).
// ---------------------------------------------------------------------------
__global__ __launch_bounds__(128) void attn_kernel(
    const h16* __restrict__ Qh, const h16* __restrict__ Kh, const h16* __restrict__ Vt,
    const unsigned char* __restrict__ mask, float* __restrict__ out)
{
  __shared__ h16 kShm[2][32 * 64];   // [key][d]  4 KB per buffer
  __shared__ h16 vShm[2][64 * 32];   // [n][key]  4 KB per buffer
  __shared__ h16 pShm[4][16 * 32];   // per-wave P staging, 1 KB each

  const int tid  = (int)threadIdx.x;
  const int lane = tid & 31;
  const int wib  = tid >> 5;
  const int b    = (int)(blockIdx.x >> 6);            // 4 batches x 64 blocks
  const int q0   = ((int)(blockIdx.x & 63)) * 64 + wib * 16;
  const int m    = lane & 15;
  const int kg   = lane >> 4;

  const h16* kgp = Kh + (size_t)b * Ss * Dd;
  const h16* vgp = Vt + (size_t)b * Dd * Ss;

  // provably-scalar lead-wave predicate (TDM ignores EXEC; need a real branch)
  const bool lead = (__builtin_amdgcn_readfirstlane(wib) == 0);

  // preload Q A-fragments (16x64 = two 16x32 chunks)
  v16h aq0, aq1;
  {
    const h16* qr = Qh + ((size_t)b * Ss + q0 + m) * Dd + kg * 16;
    #pragma unroll
    for (int j = 0; j < 8; ++j) {
      aq0[j] = qr[j];       aq0[8 + j] = qr[8 + j];
      aq1[j] = qr[32 + j];  aq1[8 + j] = qr[40 + j];
    }
  }

  const float scale   = 1.4426950408889634f * 0.125f;   // log2(e)/sqrt(64)
  const float MASKVAL = -1.4426950408889634e20f;        // -1e20 * log2(e)

  float m2[8], ls[8];
  #pragma unroll
  for (int r = 0; r < 8; ++r) { m2[r] = -3.0e38f; ls[r] = 0.0f; }
  v8f o[4] = {};

  const unsigned char* mbase = mask + ((size_t)b * Ss + q0 + 8 * kg) * (size_t)Ss + m;
  h16* pw = &pShm[wib][0];

#if USE_TDM
  if (lead) {
    tdm_load_tile_b16((unsigned)(uintptr_t)&kShm[0][0], kgp, Dd, 32, Dd, Ss, Dd);
    tdm_load_tile_b16((unsigned)(uintptr_t)&vShm[0][0], vgp, 32, Dd, Ss, Dd, Ss);
    __builtin_amdgcn_s_wait_tensorcnt(0);
  }
  __syncthreads();
#else
  // manual staging: 512 chunks of 16B over 128 threads = 2+2 per thread
  const int kr0_ = tid >> 3,         kc0_ = (tid & 7) * 8;   // K tile 32x64
  const int vr0_ = tid >> 2,         vc0_ = (tid & 3) * 8;   // V tile 64x32
  v8h ka = *(const v8h*)(kgp + (size_t)kr0_ * Dd + kc0_);
  v8h kb_ = *(const v8h*)(kgp + (size_t)(kr0_ + 16) * Dd + kc0_);
  v8h va = *(const v8h*)(vgp + (size_t)vr0_ * Ss + vc0_);
  v8h vb_ = *(const v8h*)(vgp + (size_t)(vr0_ + 32) * Ss + vc0_);
  *(v8h*)(&kShm[0][kr0_ * 64 + kc0_])        = ka;
  *(v8h*)(&kShm[0][(kr0_ + 16) * 64 + kc0_]) = kb_;
  *(v8h*)(&vShm[0][vr0_ * 32 + vc0_])        = va;
  *(v8h*)(&vShm[0][(vr0_ + 32) * 32 + vc0_]) = vb_;
  __syncthreads();
#endif
  int cur = 0;

  for (int kt = 0; kt < Ss; kt += 32) {
    const int ktn = (kt + 32 < Ss) ? kt + 32 : kt;   // next tile (dummy on last)
#if USE_TDM
    if (lead && kt + 32 < Ss) {
      const int nxt = cur ^ 1;
      tdm_load_tile_b16((unsigned)(uintptr_t)&kShm[nxt][0],
                        kgp + (size_t)ktn * Dd, Dd, 32, Dd, Ss, Dd);
      tdm_load_tile_b16((unsigned)(uintptr_t)&vShm[nxt][0],
                        vgp + ktn, 32, Dd, Ss, Dd, Ss);
    }
#else
    ka  = *(const v8h*)(kgp + (size_t)(ktn + kr0_) * Dd + kc0_);
    kb_ = *(const v8h*)(kgp + (size_t)(ktn + kr0_ + 16) * Dd + kc0_);
    va  = *(const v8h*)(vgp + (size_t)vr0_ * Ss + ktn + vc0_);
    vb_ = *(const v8h*)(vgp + (size_t)(vr0_ + 32) * Ss + ktn + vc0_);
#endif

    // ---- S = Q K^T (16 queries x 32 keys) from LDS ----
    const h16* kb = &kShm[cur][0];
    v8f s0 = {}, s1 = {};
    #pragma unroll
    for (int c = 0; c < 2; ++c) {
      const v16h bk0 = *(const v16h*)(kb + (size_t)m        * 64 + c * 32 + kg * 16);
      const v16h bk1 = *(const v16h*)(kb + (size_t)(16 + m) * 64 + c * 32 + kg * 16);
      const v16h aq  = c ? aq1 : aq0;
      s0 = __builtin_amdgcn_wmma_f32_16x16x32_f16(false, aq, false, bk0, (short)0, s0, false, false);
      s1 = __builtin_amdgcn_wmma_f32_16x16x32_f16(false, aq, false, bk1, (short)0, s1, false, false);
    }

    // ---- mask + online softmax (row reductions across 16-lane halves) ----
    #pragma unroll
    for (int r = 0; r < 8; ++r) {
      float x0 = mbase[(size_t)r * Ss + kt]      ? MASKVAL : s0[r] * scale;
      float x1 = mbase[(size_t)r * Ss + kt + 16] ? MASKVAL : s1[r] * scale;
      float t = fmaxf(x0, x1);
      t = fmaxf(t, __shfl_xor(t, 1, 32));
      t = fmaxf(t, __shfl_xor(t, 2, 32));
      t = fmaxf(t, __shfl_xor(t, 4, 32));
      t = fmaxf(t, __shfl_xor(t, 8, 32));
      float mn = fmaxf(m2[r], t);
      float cf = exp2f(m2[r] - mn);
      m2[r] = mn;
      float p0 = exp2f(x0 - mn);
      float p1 = exp2f(x1 - mn);
      float rs = p0 + p1;
      rs += __shfl_xor(rs, 1, 32);
      rs += __shfl_xor(rs, 2, 32);
      rs += __shfl_xor(rs, 4, 32);
      rs += __shfl_xor(rs, 8, 32);
      ls[r] = ls[r] * cf + rs;
      #pragma unroll
      for (int n4 = 0; n4 < 4; ++n4) o[n4][r] *= cf;
      pw[(r + 8 * kg) * 32 + m]      = (h16)p0;
      pw[(r + 8 * kg) * 32 + 16 + m] = (h16)p1;
    }

    // ---- reload P as A fragment (same-wave LDS: in order) ----
    v16h ap;
    {
      const h16* pr = pw + m * 32 + kg * 16;
      #pragma unroll
      for (int j = 0; j < 8; ++j) { ap[j] = pr[j]; ap[8 + j] = pr[8 + j]; }
    }

    // ---- O += P V from LDS V tile ([n][key] -> contiguous per lane) ----
    {
      const h16* vb = &vShm[cur][0];
      #pragma unroll
      for (int n4 = 0; n4 < 4; ++n4) {
        const v16h bv = *(const v16h*)(vb + (size_t)(n4 * 16 + m) * 32 + kg * 16);
        o[n4] = __builtin_amdgcn_wmma_f32_16x16x32_f16(
            false, ap, false, bv, (short)0, o[n4], false, false);
      }
    }

    // ---- complete next-tile staging, single barrier ----
#if USE_TDM
    if (lead) __builtin_amdgcn_s_wait_tensorcnt(0);
#else
    *(v8h*)(&kShm[cur ^ 1][kr0_ * 64 + kc0_])        = ka;
    *(v8h*)(&kShm[cur ^ 1][(kr0_ + 16) * 64 + kc0_]) = kb_;
    *(v8h*)(&vShm[cur ^ 1][vr0_ * 32 + vc0_])        = va;
    *(v8h*)(&vShm[cur ^ 1][(vr0_ + 32) * 32 + vc0_]) = vb_;
#endif
    __syncthreads();
    cur ^= 1;
  }

  // ---- finalize: divide by row sums, store fp32 ----
  #pragma unroll
  for (int r = 0; r < 8; ++r) {
    float inv = 1.0f / ls[r];
    float* orow = out + ((size_t)b * Ss + q0 + r + 8 * kg) * Dd + m;
    #pragma unroll
    for (int n4 = 0; n4 < 4; ++n4) orow[n4 * 16] = o[n4][r] * inv;
  }
}

// ---------------------------------------------------------------------------
// Inputs (setup_inputs order): queries, keys, values, mask(bool), W_q, W_k, W_v
// Workspace: f16 Qh | Kh | Vt | Wh  =  2+2+2 MB + 384 KB.
// ---------------------------------------------------------------------------
extern "C" void kernel_launch(void* const* d_in, const int* in_sizes, int n_in,
                              void* d_out, int out_size, void* d_ws, size_t ws_size,
                              hipStream_t stream)
{
  (void)in_sizes; (void)n_in; (void)out_size; (void)ws_size;
  const float* q  = (const float*)d_in[0];
  const float* k  = (const float*)d_in[1];
  const float* v  = (const float*)d_in[2];
  const unsigned char* mask = (const unsigned char*)d_in[3]; // jnp bool: 1 byte/elt
  const float* wq = (const float*)d_in[4];
  const float* wk = (const float*)d_in[5];
  const float* wv = (const float*)d_in[6];

  h16* Qh = (h16*)d_ws;
  h16* Kh = Qh + (size_t)Bb * Ss * Dd;
  h16* Vt = Kh + (size_t)Bb * Ss * Dd;
  h16* Wh = Vt + (size_t)Bb * Ss * Dd;

  cvt_w_kernel<<<768, 256, 0, stream>>>(wq, wk, wv, Wh);
  // 3 matrices * 1024 tiles = 3072 waves, 8 waves per block
  proj_kernel<<<384, 256, 0, stream>>>(q, k, v, Wh, Qh, Kh, Vt);
  // 4 batches * 64 blocks, each block = 4 waves = 64 query rows
  attn_kernel<<<256, 128, 0, stream>>>(Qh, Kh, Vt, mask, (float*)d_out);
}
